// Model_28991029248765
// MI455X (gfx1250) — matrix operand order repaired and checked by reading, not compiled, not run on hardware
//
#include <hip/hip_runtime.h>
#include <cstdint>

// ---------------- constants ----------------
#define DS_C   2048
#define TS_C   4096
#define NN     6144          // DS + TS, also feature dim of x
#define EE     1048576       // directed edges
#define EPc    (EE + NN)     // edges + self loops
#define NHEAD  4
#define GAMMA_C 0.03125f     // 2^-5
#define NEG_SLOPE 0.2f

typedef __attribute__((ext_vector_type(16))) __bf16 v16bf;
typedef __attribute__((ext_vector_type(8)))  __bf16 v8bf;
typedef __attribute__((ext_vector_type(8)))  float  v8f;
typedef __bf16 bf16_t;

// ---------------- helpers ----------------
__device__ __forceinline__ bf16_t f2bf(float f) {
  unsigned u = __float_as_uint(f);
  u += 0x7FFFu + ((u >> 16) & 1u);           // round to nearest even
  unsigned short s = (unsigned short)(u >> 16);
  return __builtin_bit_cast(bf16_t, s);
}

__device__ __forceinline__ void edge_nodes(const int* __restrict__ ei, int e, int& src, int& dst) {
  if (e < EE) { src = ei[e]; dst = ei[EE + e]; }
  else        { src = e - EE; dst = e - EE; }
}

// ---------------- elementwise ----------------
__global__ void k_f32_to_bf16(const float* __restrict__ src, bf16_t* __restrict__ dst, long long n) {
  long long i = (long long)blockIdx.x * blockDim.x + threadIdx.x;
  if (i < n) dst[i] = f2bf(src[i]);
}

// src: rows x cols (f32, row-major) -> dst: cols x rows (bf16, row-major)
__global__ void k_f32_to_bf16_T(const float* __restrict__ src, bf16_t* __restrict__ dst,
                                int rows, int cols) {
  long long i = (long long)blockIdx.x * blockDim.x + threadIdx.x;
  if (i >= (long long)rows * cols) return;
  int r = (int)(i / cols), c = (int)(i % cols);
  dst[(long long)c * rows + r] = f2bf(src[i]);
}

__global__ void k_scale_copy(const float* __restrict__ src, float* __restrict__ dst, float s, long long n) {
  long long i = (long long)blockIdx.x * blockDim.x + threadIdx.x;
  if (i < n) dst[i] = s * src[i];
}

__global__ void k_set_u32(unsigned* p, unsigned v) { *p = v; }

// ---------------- WMMA bf16 GEMM (NT) ----------------
// C[M,N] (f32, ldc) = A[M,K] (bf16 row-major, lda) * B[N,K]^T (bf16 row-major, ldb)
// grid = (N/64, M/64), block = 32 (one wave). One wave computes a 64x64 C tile.
// launch_bounds(32,1): full VGPR budget, no spills.
// WMMA issue order is i-outer/j-inner so b0..b3 are all live across the whole
// 16-WMMA block: the allocator must keep them in distinct registers and the
// scheduler can cluster all 16 fragment loads up front with staggered waits.
__device__ __forceinline__ v16bf load_afrag(const bf16_t* p) {
  v8bf lo = *(const v8bf*)(p);
  v8bf hi = *(const v8bf*)(p + 8);
  return __builtin_shufflevector(lo, hi, 0, 1, 2, 3, 4, 5, 6, 7,
                                 8, 9, 10, 11, 12, 13, 14, 15);
}

__global__ void __launch_bounds__(32, 1)
k_gemm_nt(const bf16_t* __restrict__ A, const bf16_t* __restrict__ B,
          float* __restrict__ C, int K, int lda, int ldb, int ldc) {
  const int lane = threadIdx.x;          // 0..31
  const int h    = lane >> 4;            // lane-half
  const int mr   = lane & 15;
  const long long row0 = (long long)blockIdx.y * 64;
  const long long col0 = (long long)blockIdx.x * 64;

  // per-row pointers, advanced by 32 elements (64B) per K-step
  const bf16_t* Ap0 = A + (row0 + mr) * lda + 16 * h;
  const bf16_t* Ap1 = Ap0 + 16 * (long long)lda;
  const bf16_t* Ap2 = Ap0 + 32 * (long long)lda;
  const bf16_t* Ap3 = Ap0 + 48 * (long long)lda;
  const bf16_t* Bp0 = B + (col0 + mr) * ldb + 16 * h;
  const bf16_t* Bp1 = Bp0 + 16 * (long long)ldb;
  const bf16_t* Bp2 = Bp0 + 32 * (long long)ldb;
  const bf16_t* Bp3 = Bp0 + 48 * (long long)ldb;

  v8f acc[4][4];
#pragma unroll
  for (int i = 0; i < 4; ++i)
#pragma unroll
    for (int j = 0; j < 4; ++j) acc[i][j] = v8f{};

  for (int k0 = 0; k0 < K; k0 += 32) {
    if (k0 + 32 < K) {   // hint next K-slice into cache (global_prefetch_b8)
      __builtin_prefetch(Ap0 + 32, 0, 3);
      __builtin_prefetch(Bp0 + 32, 0, 3);
    }
    // issue ALL fragment loads up front (16x global_load_b128)
    v16bf b0 = *(const v16bf*)(Bp0); Bp0 += 32;   // 32B contiguous -> 2x b128
    v16bf b1 = *(const v16bf*)(Bp1); Bp1 += 32;
    v16bf b2 = *(const v16bf*)(Bp2); Bp2 += 32;
    v16bf b3 = *(const v16bf*)(Bp3); Bp3 += 32;
    v16bf a0 = load_afrag(Ap0); Ap0 += 32;
    v16bf a1 = load_afrag(Ap1); Ap1 += 32;
    v16bf a2 = load_afrag(Ap2); Ap2 += 32;
    v16bf a3 = load_afrag(Ap3); Ap3 += 32;

    // i-outer / j-inner: first 4 WMMAs touch b0..b3 -> all B frags live
    acc[0][0] = __builtin_amdgcn_wmma_f32_16x16x32_bf16(false, a0, false, b0, (short)0, acc[0][0], false, false);
    acc[0][1] = __builtin_amdgcn_wmma_f32_16x16x32_bf16(false, a0, false, b1, (short)0, acc[0][1], false, false);
    acc[0][2] = __builtin_amdgcn_wmma_f32_16x16x32_bf16(false, a0, false, b2, (short)0, acc[0][2], false, false);
    acc[0][3] = __builtin_amdgcn_wmma_f32_16x16x32_bf16(false, a0, false, b3, (short)0, acc[0][3], false, false);

    acc[1][0] = __builtin_amdgcn_wmma_f32_16x16x32_bf16(false, a1, false, b0, (short)0, acc[1][0], false, false);
    acc[1][1] = __builtin_amdgcn_wmma_f32_16x16x32_bf16(false, a1, false, b1, (short)0, acc[1][1], false, false);
    acc[1][2] = __builtin_amdgcn_wmma_f32_16x16x32_bf16(false, a1, false, b2, (short)0, acc[1][2], false, false);
    acc[1][3] = __builtin_amdgcn_wmma_f32_16x16x32_bf16(false, a1, false, b3, (short)0, acc[1][3], false, false);

    acc[2][0] = __builtin_amdgcn_wmma_f32_16x16x32_bf16(false, a2, false, b0, (short)0, acc[2][0], false, false);
    acc[2][1] = __builtin_amdgcn_wmma_f32_16x16x32_bf16(false, a2, false, b1, (short)0, acc[2][1], false, false);
    acc[2][2] = __builtin_amdgcn_wmma_f32_16x16x32_bf16(false, a2, false, b2, (short)0, acc[2][2], false, false);
    acc[2][3] = __builtin_amdgcn_wmma_f32_16x16x32_bf16(false, a2, false, b3, (short)0, acc[2][3], false, false);

    acc[3][0] = __builtin_amdgcn_wmma_f32_16x16x32_bf16(false, a3, false, b0, (short)0, acc[3][0], false, false);
    acc[3][1] = __builtin_amdgcn_wmma_f32_16x16x32_bf16(false, a3, false, b1, (short)0, acc[3][1], false, false);
    acc[3][2] = __builtin_amdgcn_wmma_f32_16x16x32_bf16(false, a3, false, b2, (short)0, acc[3][2], false, false);
    acc[3][3] = __builtin_amdgcn_wmma_f32_16x16x32_bf16(false, a3, false, b3, (short)0, acc[3][3], false, false);
  }

  // C/D layout: VGPR r holds M = r + 8h, N = lane%16
#pragma unroll
  for (int i = 0; i < 4; ++i) {
#pragma unroll
    for (int j = 0; j < 4; ++j) {
      float* Cp = C + (row0 + i * 16 + 8 * h) * ldc + col0 + j * 16 + mr;
#pragma unroll
      for (int r = 0; r < 8; ++r) Cp[(long long)r * ldc] = acc[i][j][r];
    }
  }
}

// ---------------- GAT attention ----------------
__global__ void k_scores(const float* __restrict__ z, const float* __restrict__ asrc,
                         const float* __restrict__ adst, float* __restrict__ ssrc,
                         float* __restrict__ sdst, int F) {
  int i = blockIdx.x * blockDim.x + threadIdx.x;
  if (i >= NN * NHEAD) return;
  int node = i / NHEAD, h = i % NHEAD;
  const float* zp = z + (long long)node * NHEAD * F + (long long)h * F;
  const float* ap = asrc + h * F;
  const float* dp = adst + h * F;
  float ss = 0.f, sd = 0.f;
  for (int f = 0; f < F; ++f) { float v = zp[f]; ss += v * ap[f]; sd += v * dp[f]; }
  ssrc[i] = ss; sdst[i] = sd;
}

__device__ __forceinline__ float edge_logit(const int* ei, const float* ssrc, const float* sdst,
                                            int e, int h, int& src, int& dst) {
  edge_nodes(ei, e, src, dst);
  float lg = ssrc[src * NHEAD + h] + sdst[dst * NHEAD + h];
  return lg < 0.f ? NEG_SLOPE * lg : lg;
}

__global__ void k_edge_max(const int* __restrict__ ei, const float* __restrict__ ssrc,
                           const float* __restrict__ sdst, unsigned* __restrict__ segmax) {
  int idx = blockIdx.x * blockDim.x + threadIdx.x;
  if (idx >= EPc * NHEAD) return;
  int e = idx / NHEAD, h = idx % NHEAD;
  int src, dst;
  float lg = edge_logit(ei, ssrc, sdst, e, h, src, dst);
  unsigned u = __float_as_uint(lg);
  u = (u & 0x80000000u) ? ~u : (u | 0x80000000u);   // order-preserving encode
  atomicMax(&segmax[dst * NHEAD + h], u);
}

__global__ void k_edge_exp(const int* __restrict__ ei, const float* __restrict__ ssrc,
                           const float* __restrict__ sdst, const unsigned* __restrict__ segmax,
                           float* __restrict__ ae, float* __restrict__ segsum) {
  int idx = blockIdx.x * blockDim.x + threadIdx.x;
  if (idx >= EPc * NHEAD) return;
  int e = idx / NHEAD, h = idx % NHEAD;
  int src, dst;
  float lg = edge_logit(ei, ssrc, sdst, e, h, src, dst);
  unsigned enc = segmax[dst * NHEAD + h];
  unsigned bits = (enc & 0x80000000u) ? (enc & 0x7FFFFFFFu) : ~enc;
  float m = __uint_as_float(bits);
  float ev = __expf(lg - m);
  ae[idx] = ev;
  atomicAdd(&segsum[dst * NHEAD + h], ev);
}

__global__ void k_edge_norm(const int* __restrict__ ei, const float* __restrict__ segsum,
                            float* __restrict__ ae) {
  int idx = blockIdx.x * blockDim.x + threadIdx.x;
  if (idx >= EPc * NHEAD) return;
  int e = idx / NHEAD, h = idx % NHEAD;
  int src, dst;
  edge_nodes(ei, e, src, dst);
  ae[idx] = ae[idx] / segsum[dst * NHEAD + h];
}

__global__ void k_scatter(const int* __restrict__ ei, const float* __restrict__ z,
                          const float* __restrict__ ae, float* __restrict__ agg,
                          int C, int F, long long total) {
  long long idx = (long long)blockIdx.x * blockDim.x + threadIdx.x;
  if (idx >= total) return;
  int e = (int)(idx / C);
  int c = (int)(idx % C);
  int h = c / F;
  int src, dst;
  edge_nodes(ei, e, src, dst);
  float w = ae[(long long)e * NHEAD + h];
  atomicAdd(&agg[(long long)dst * C + c], w * z[(long long)src * C + c]);
}

__global__ void k_bias_relu(const float* __restrict__ agg, const float* __restrict__ b,
                            float* __restrict__ H, bf16_t* __restrict__ Hb, int C) {
  long long idx = (long long)blockIdx.x * blockDim.x + threadIdx.x;
  if (idx >= (long long)NN * C) return;
  int c = (int)(idx % C);
  float v = agg[idx] + b[c];
  v = v > 0.f ? v : 0.f;
  H[idx] = v;
  if (Hb) Hb[idx] = f2bf(v);
}

// ---------------- GIP kernel pieces ----------------
__global__ void k_row_minmax(const float* __restrict__ Y, bf16_t* __restrict__ Yb, int F) {
  __shared__ float smn[128], smx[128];
  int row = blockIdx.x, t = threadIdx.x;
  const float* yr = Y + (long long)row * F;
  float mn = 3.4e38f, mx = -3.4e38f;
  for (int f = t; f < F; f += 128) { float v = yr[f]; mn = fminf(mn, v); mx = fmaxf(mx, v); }
  smn[t] = mn; smx[t] = mx;
  __syncthreads();
  for (int s = 64; s > 0; s >>= 1) {
    if (t < s) { smn[t] = fminf(smn[t], smn[t + s]); smx[t] = fmaxf(smx[t], smx[t + s]); }
    __syncthreads();
  }
  float base = smn[0];
  float rng  = smx[0] - base;
  float inv  = (rng == 0.f) ? 1.f : 1.f / rng;
  for (int f = t; f < F; f += 128) Yb[(long long)row * F + f] = f2bf((yr[f] - base) * inv);
}

__global__ void k_diag_sum(const float* __restrict__ G, int R, float* __restrict__ s) {
  int i = blockIdx.x * blockDim.x + threadIdx.x;
  if (i < R) atomicAdd(s, G[(long long)i * R + i]);
}

__global__ void k_gip_acc(const float* __restrict__ G, int R, const float* __restrict__ ssum,
                          float wgt, float* __restrict__ acc) {
  long long idx = (long long)blockIdx.x * blockDim.x + threadIdx.x;
  if (idx >= (long long)R * R) return;
  int i = (int)(idx / R), j = (int)(idx % R);
  float invmd = (float)R / ssum[0];            // 1/meanDiag
  float k  = G[idx] * invmd;
  float di = G[(long long)i * R + i] * invmd;
  float dj = G[(long long)j * R + j] * invmd;
  acc[idx] += wgt * __expf(-GAMMA_C * (di + dj - 2.f * k));
}

__global__ void k_min_nonzero(const float* __restrict__ acc, long long n, unsigned* __restrict__ mv) {
  long long idx = (long long)blockIdx.x * blockDim.x + threadIdx.x;
  if (idx >= n) return;
  float v = fabsf(acc[idx]);
  if (v != 0.f) atomicMin(mv, __float_as_uint(v));   // positive float bits are monotonic
}

__global__ void k_norm_div(const float* __restrict__ acc, const unsigned* __restrict__ mv,
                           int R, bf16_t* __restrict__ Kb) {
  long long idx = (long long)blockIdx.x * blockDim.x + threadIdx.x;
  if (idx >= (long long)R * R) return;
  int j = (int)(idx % R);
  float minv = __uint_as_float(*mv);
  float v = fabsf(acc[idx]);                  if (v == 0.f) v = minv;
  float d = fabsf(acc[(long long)j * R + j]); if (d == 0.f) d = minv;
  Kb[idx] = f2bf(v / d);
}

__global__ void k_combine(const float* __restrict__ P1, const float* __restrict__ P2,
                          float* __restrict__ out) {
  long long idx = (long long)blockIdx.x * blockDim.x + threadIdx.x;
  if (idx >= (long long)DS_C * TS_C) return;
  int i = (int)(idx / TS_C), j = (int)(idx % TS_C);
  out[idx] = 0.5f * (P1[idx] + P2[(long long)j * DS_C + i]);
}

// ---------------- host orchestration ----------------
extern "C" void kernel_launch(void* const* d_in, const int* in_sizes, int n_in,
                              void* d_out, int out_size, void* d_ws, size_t ws_size,
                              hipStream_t stream) {
  const float* x        = (const float*)d_in[0];
  const int*   ei       = (const int*)d_in[1];
  const float* W[4]     = {(const float*)d_in[2],  (const float*)d_in[6],
                           (const float*)d_in[10], (const float*)d_in[14]};
  const float* asrc[4]  = {(const float*)d_in[3],  (const float*)d_in[7],
                           (const float*)d_in[11], (const float*)d_in[15]};
  const float* adst[4]  = {(const float*)d_in[4],  (const float*)d_in[8],
                           (const float*)d_in[12], (const float*)d_in[16]};
  const float* bias[4]  = {(const float*)d_in[5],  (const float*)d_in[9],
                           (const float*)d_in[13], (const float*)d_in[17]};
  const float* alpha1   = (const float*)d_in[18];
  const float* alpha2   = (const float*)d_in[19];
  const float* drug_sim = (const float*)d_in[20];
  const float* tgt_sim  = (const float*)d_in[21];
  float* out = (float*)d_out;

  // bump allocator over d_ws
  char* base = (char*)d_ws;
  size_t off = 0;
  auto alloc = [&](size_t bytes) -> void* {
    off = (off + 255) & ~(size_t)255;
    void* p = base + off;
    off += bytes;
    return p;
  };

  bf16_t* xbf  = (bf16_t*)alloc((size_t)NN * NN * 2);
  bf16_t* wT[4];                                       // W^T : C x K (bf16)
  wT[0] = (bf16_t*)alloc((size_t)256 * NN * 2);
  wT[1] = (bf16_t*)alloc((size_t)256 * 256 * 2);
  wT[2] = (bf16_t*)alloc((size_t)256 * 256 * 2);
  wT[3] = (bf16_t*)alloc((size_t)128 * 256 * 2);
  bf16_t* a1T  = (bf16_t*)alloc((size_t)TS_C * DS_C * 2);   // alpha1^T : TS x DS
  bf16_t* a2T  = (bf16_t*)alloc((size_t)DS_C * TS_C * 2);   // alpha2^T : DS x TS
  float*  z    = (float*)alloc((size_t)NN * 256 * 4);
  float*  ssrc = (float*)alloc((size_t)NN * NHEAD * 4);
  float*  sdst = (float*)alloc((size_t)NN * NHEAD * 4);
  unsigned* segmax = (unsigned*)alloc((size_t)NN * NHEAD * 4);
  float*  segsum = (float*)alloc((size_t)NN * NHEAD * 4);
  float*  ae   = (float*)alloc((size_t)EPc * NHEAD * 4);
  float*  agg  = (float*)alloc((size_t)NN * 256 * 4);
  float*  Hm[4];
  Hm[0] = (float*)alloc((size_t)NN * 256 * 4);
  Hm[1] = (float*)alloc((size_t)NN * 256 * 4);
  Hm[2] = (float*)alloc((size_t)NN * 256 * 4);
  Hm[3] = (float*)alloc((size_t)NN * 128 * 4);
  bf16_t* Hbf = (bf16_t*)alloc((size_t)NN * 256 * 2);
  bf16_t* ybf = (bf16_t*)alloc((size_t)TS_C * 256 * 2);
  float*  G    = (float*)alloc((size_t)TS_C * TS_C * 4);
  float*  accd = (float*)alloc((size_t)DS_C * DS_C * 4);
  float*  acct = (float*)alloc((size_t)TS_C * TS_C * 4);
  bf16_t* kdbf = (bf16_t*)alloc((size_t)DS_C * DS_C * 2);
  bf16_t* ktbf = (bf16_t*)alloc((size_t)TS_C * TS_C * 2);
  float*    ssum = (float*)alloc(256);
  unsigned* minv = (unsigned*)alloc(256);
  // aliases (lifetimes disjoint): P1 over G, P2 over xbf
  float* P1 = G;
  float* P2 = (float*)xbf;
  (void)ws_size; (void)n_in; (void)in_sizes; (void)out_size;

  auto cvt = [&](const float* s, bf16_t* d, long long n) {
    k_f32_to_bf16<<<(unsigned)((n + 255) / 256), 256, 0, stream>>>(s, d, n);
  };
  auto cvtT = [&](const float* s, bf16_t* d, int rows, int cols) {
    long long n = (long long)rows * cols;
    k_f32_to_bf16_T<<<(unsigned)((n + 255) / 256), 256, 0, stream>>>(s, d, rows, cols);
  };

  // ---- conversions (B operands transposed to N x K for the NT GEMM) ----
  cvt(x, xbf, (long long)NN * NN);
  cvtT(W[0], wT[0], NN, 256);
  cvtT(W[1], wT[1], 256, 256);
  cvtT(W[2], wT[2], 256, 256);
  cvtT(W[3], wT[3], 256, 128);
  cvtT(alpha1, a1T, DS_C, TS_C);
  cvtT(alpha2, a2T, TS_C, DS_C);

  // ---- GAT layers ----
  auto run_layer = [&](const bf16_t* hin, int Kdim, int L, int C, float* Hout, bf16_t* Hob) {
    int F = C / NHEAD;
    hipMemsetAsync(agg, 0, (size_t)NN * C * 4, stream);
    hipMemsetAsync(segmax, 0, (size_t)NN * NHEAD * 4, stream);
    hipMemsetAsync(segsum, 0, (size_t)NN * NHEAD * 4, stream);
    // z = hin @ W = hin (NN x K) * (W^T)^T ; W^T is C x K
    k_gemm_nt<<<dim3(C / 64, NN / 64), 32, 0, stream>>>(hin, wT[L], z, Kdim, Kdim, Kdim, C);
    int nh = NN * NHEAD;
    k_scores<<<(nh + 255) / 256, 256, 0, stream>>>(z, asrc[L], adst[L], ssrc, sdst, F);
    int eh = EPc * NHEAD;
    k_edge_max<<<(eh + 255) / 256, 256, 0, stream>>>(ei, ssrc, sdst, segmax);
    k_edge_exp<<<(eh + 255) / 256, 256, 0, stream>>>(ei, ssrc, sdst, segmax, ae, segsum);
    k_edge_norm<<<(eh + 255) / 256, 256, 0, stream>>>(ei, segsum, ae);
    long long tot = (long long)EPc * C;
    k_scatter<<<(unsigned)((tot + 255) / 256), 256, 0, stream>>>(ei, z, ae, agg, C, F, tot);
    long long nc = (long long)NN * C;
    k_bias_relu<<<(unsigned)((nc + 255) / 256), 256, 0, stream>>>(agg, bias[L], Hout, Hob, C);
  };

  run_layer(xbf, NN, 0, 256, Hm[0], Hbf);
  run_layer(Hbf, 256, 1, 256, Hm[1], Hbf);   // Hbf read by GEMM before overwrite (stream order)
  run_layer(Hbf, 256, 2, 256, Hm[2], Hbf);
  run_layer(Hbf, 256, 3, 128, Hm[3], nullptr);

  // ---- GIP kernels ----
  k_scale_copy<<<(unsigned)(((long long)DS_C * DS_C + 255) / 256), 256, 0, stream>>>(
      drug_sim, accd, 0.2f, (long long)DS_C * DS_C);
  k_scale_copy<<<(unsigned)(((long long)TS_C * TS_C + 255) / 256), 256, 0, stream>>>(
      tgt_sim, acct, 0.2f, (long long)TS_C * TS_C);

  auto run_gip = [&](const float* Hmat, int C, int R, int rowoff, float wgt, float* acc) {
    k_row_minmax<<<R, 128, 0, stream>>>(Hmat + (long long)rowoff * C, ybf, C);
    // G = y @ y^T  (natural NT shape)
    k_gemm_nt<<<dim3(R / 64, R / 64), 32, 0, stream>>>(ybf, ybf, G, C, C, C, R);
    hipMemsetAsync(ssum, 0, sizeof(float), stream);
    k_diag_sum<<<(R + 255) / 256, 256, 0, stream>>>(G, R, ssum);
    unsigned blocks = (unsigned)(((long long)R * R + 255) / 256);
    k_gip_acc<<<blocks, 256, 0, stream>>>(G, R, ssum, wgt, acc);
  };

  const float PS[4] = {1.0f, 0.5f, 0.333f, 0.25f};
  const int   Cs[4] = {256, 256, 256, 128};
  for (int l = 0; l < 4; ++l) run_gip(Hm[l], Cs[l], DS_C, 0, PS[l], accd);
  for (int l = 0; l < 4; ++l) run_gip(Hm[l], Cs[l], TS_C, DS_C, PS[l], acct);

  // ---- normalized_kernel ----
  auto run_norm = [&](const float* acc, int R, bf16_t* Kb) {
    k_set_u32<<<1, 1, 0, stream>>>(minv, 0x7F800000u);   // +inf
    long long n = (long long)R * R;
    unsigned blocks = (unsigned)((n + 255) / 256);
    k_min_nonzero<<<blocks, 256, 0, stream>>>(acc, n, minv);
    k_norm_div<<<blocks, 256, 0, stream>>>(acc, minv, R, Kb);
  };
  run_norm(accd, DS_C, kdbf);
  run_norm(acct, TS_C, ktbf);

  // ---- final: (drug_k@alpha1 + (target_k@alpha2)^T)/2 ----
  // P1 = drug_k (DSxDS) * alpha1 = drug_k * (a1T)^T ; a1T is TS x DS
  k_gemm_nt<<<dim3(TS_C / 64, DS_C / 64), 32, 0, stream>>>(
      kdbf, a1T, P1, DS_C, DS_C, DS_C, TS_C);
  // P2 = target_k (TSxTS) * alpha2 = target_k * (a2T)^T ; a2T is DS x TS
  k_gemm_nt<<<dim3(DS_C / 64, TS_C / 64), 32, 0, stream>>>(
      ktbf, a2T, P2, TS_C, TS_C, TS_C, DS_C);
  long long no = (long long)DS_C * TS_C;
  k_combine<<<(unsigned)((no + 255) / 256), 256, 0, stream>>>(P1, P2, out);
}
